// MP_network_28295244546512
// MI455X (gfx1250) — compile-verified, matrix operand order checked by reading, not run on hardware
//
#include <hip/hip_runtime.h>
#include <hip/hip_bf16.h>

// ---------------------------------------------------------------------------
// Types for CDNA5 WMMA (wave32): A/B = 16 x bf16 (8 VGPRs), C/D = 8 x f32.
// ---------------------------------------------------------------------------
typedef __attribute__((ext_vector_type(16))) __bf16 v16bf;
typedef __attribute__((ext_vector_type(8)))  float  v8f;

#define EMB   600
#define KPAD  608   // 600 padded to a multiple of 32 (19 k-steps)
#define NND   50000
#define NED   100000
#define NGR   1024

__device__ __forceinline__ unsigned short f2bf(float x) {
    // round-to-nearest-even f32 -> bf16 bits
    unsigned int u = __float_as_uint(x);
    u += 0x7FFFu + ((u >> 16) & 1u);
    return (unsigned short)(u >> 16);
}

// ---------------------------------------------------------------------------
// fill
// ---------------------------------------------------------------------------
__global__ void fill_f32(float* __restrict__ p, long long n, float v) {
    long long i = (long long)blockIdx.x * blockDim.x + threadIdx.x;
    long long s = (long long)gridDim.x * blockDim.x;
    for (; i < n; i += s) p[i] = v;
}

// ---------------------------------------------------------------------------
// Pack fp32 weight matrix [Nw, K] row-major into bf16 [Nrows, Kpad],
// zero-padded in both dims so WMMA B-fragments are 16B-aligned b128 loads.
// ---------------------------------------------------------------------------
__global__ void pack_weights(const float* __restrict__ W, int Nw, int K,
                             unsigned short* __restrict__ out, int Nrows, int Kpad) {
    int i = blockIdx.x * blockDim.x + threadIdx.x;
    int total = Nrows * Kpad;
    if (i >= total) return;
    int r = i / Kpad, c = i - r * Kpad;
    float v = (r < Nw && c < K) ? W[(size_t)r * K + c] : 0.0f;
    out[i] = f2bf(v);
}

// ---------------------------------------------------------------------------
// Step 1: node embedding = sum of 5 table lookups. One block per node.
// ---------------------------------------------------------------------------
__global__ void node_embed(const int* __restrict__ atom,
                           const float* __restrict__ Wa, const float* __restrict__ Wh,
                           const float* __restrict__ Wr, const float* __restrict__ Wc,
                           const float* __restrict__ Wq, float* __restrict__ node) {
    int n = blockIdx.x;
    const float* a = Wa + (size_t)atom[n * 5 + 0] * EMB;
    const float* h = Wh + (size_t)atom[n * 5 + 1] * EMB;
    const float* r = Wr + (size_t)atom[n * 5 + 2] * EMB;
    const float* c = Wc + (size_t)atom[n * 5 + 3] * EMB;
    const float* q = Wq + (size_t)atom[n * 5 + 4] * EMB;
    float* o = node + (size_t)n * EMB;
    for (int f = threadIdx.x; f < EMB; f += blockDim.x)
        o[f] = a[f] + h[f] + r[f] + c[f] + q[f];
}

// ---------------------------------------------------------------------------
// Steps 2+3: one message-passing layer over E bonds + N self-loops.
// Edge embeddings are recomputed from the (L2-resident) tables on the fly.
// One block per edge; scatter-add via global_atomic_add_f32.
// ---------------------------------------------------------------------------
__global__ void mp_layer(const float* __restrict__ node_in,
                         float* __restrict__ node_out,
                         const int* __restrict__ bond_index,  // [2, E]
                         const int* __restrict__ bond_attr,   // [E, 2]
                         const int* __restrict__ atom,        // [N, 5]
                         const float* __restrict__ W_bond_type,
                         const float* __restrict__ W_bond_atom_i,
                         const float* __restrict__ W_sl_atom,
                         const float* __restrict__ W_sl) {
    int e = blockIdx.x;
    int src, dst;
    const float *t0, *t1;
    if (e < NED) {
        src = bond_index[e];
        dst = bond_index[NED + e];
        t0  = W_bond_type   + (size_t)bond_attr[e * 2 + 0] * EMB;
        t1  = W_bond_atom_i + (size_t)bond_attr[e * 2 + 1] * EMB;
    } else {
        int n = e - NED;             // self-loop
        src = dst = n;
        t0  = W_sl_atom + (size_t)atom[n * 5 + 0] * EMB;
        t1  = W_sl;
    }
    const float* xin  = node_in  + (size_t)src * EMB;
    float*       xout = node_out + (size_t)dst * EMB;
    for (int f = threadIdx.x; f < EMB; f += blockDim.x) {
        float m = xin[f] * (t0[f] + t1[f]);
        atomicAdd(&xout[f], m);
    }
}

// ---------------------------------------------------------------------------
// GEMM with fused bias + ReLU:  out[M,Nout] = relu( act(A[M,K]) @ Wp^T + b )
//   A:  fp32, converted to bf16 while staged into LDS (relu_input applies it)
//   Wp: bf16, pre-packed [ntiles*16, KPAD] row-major (zero padded)
// Block = 256 threads = 8 waves; block owns one 16-row M tile; wave w owns
// N-tiles {w, w+8, ...}. Inner op: v_wmma_f32_16x16x32_bf16.
//
// Fragment layout per CDNA5 ISA (16-bit A / B, wave32):
//   lanes 0..15 : row/col = lane,    K = {kb..kb+7, kb+16..kb+23}
//   lanes 16..31: row/col = lane-16, K = {kb+8..kb+15, kb+24..kb+31}
// C/D: lanes 0..15 N=lane M=r(0..7); lanes 16..31 N=lane-16 M=8+r.
// ---------------------------------------------------------------------------
template <int NTJ>
__global__ void gemm_bias_relu(const float* __restrict__ A, int M, int K,
                               const unsigned short* __restrict__ Wp,
                               const float* __restrict__ bias,
                               float* __restrict__ out, int Nout, int ntiles,
                               int relu_input) {
    __shared__ alignas(16) unsigned short ldsA[16 * 32];  // 16 rows x 32 k, bf16

    const int tid   = threadIdx.x;
    const int lane  = tid & 31;     // wave32 (gfx1250 is wave32-only)
    const int wave  = tid >> 5;     // 0..7
    const int mbase = blockIdx.x * 16;

    v8f acc[NTJ] = {};

    const int arow = lane & 15;
    const int koff = (lane < 16) ? 0 : 8;
    const int ksteps = (K + 31) / 32;

    for (int ks = 0; ks < ksteps; ++ks) {
        const int kbase = ks * 32;
        __syncthreads();  // protect LDS reads of previous iteration
        // stage 16x32 fp32 A tile -> bf16 LDS (512 elems, 2 per thread)
        for (int i = tid; i < 512; i += 256) {
            int r = i >> 5, c = i & 31;
            int m = mbase + r, k = kbase + c;
            float v = 0.0f;
            if (m < M && k < K) v = A[(size_t)m * K + k];
            if (relu_input) v = fmaxf(v, 0.0f);
            ldsA[r * 32 + c] = f2bf(v);
        }
        __syncthreads();

        // A fragment from LDS: two aligned 16B reads (ds_load_b128)
        union { v16bf v; uint4 u[2]; } af;
        af.u[0] = *(const uint4*)&ldsA[arow * 32 + koff];
        af.u[1] = *(const uint4*)&ldsA[arow * 32 + koff + 16];

        for (int j = 0; j < NTJ; ++j) {
            int t = wave + 8 * j;
            if (t >= ntiles) break;
            int n = t * 16 + (lane & 15);
            const unsigned short* wrow = Wp + (size_t)n * KPAD + kbase + koff;
            union { v16bf v; uint4 u[2]; } bf;
            bf.u[0] = *(const uint4*)(wrow);        // global_load_b128
            bf.u[1] = *(const uint4*)(wrow + 16);
            acc[j] = __builtin_amdgcn_wmma_f32_16x16x32_bf16(
                false, af.v, false, bf.v, (short)0, acc[j], false, false);
        }
    }

    // epilogue: bias + ReLU, store per documented C/D layout
    const int mrow0 = (lane < 16) ? 0 : 8;
    for (int j = 0; j < NTJ; ++j) {
        int t = wave + 8 * j;
        if (t >= ntiles) break;
        int n = t * 16 + (lane & 15);
        if (n >= Nout) continue;
        float b = bias ? bias[n] : 0.0f;
        for (int r = 0; r < 8; ++r) {
            int m = mbase + mrow0 + r;
            if (m >= M) continue;
            float v = fmaxf(acc[j][r] + b, 0.0f);
            out[(size_t)m * Nout + n] = v;
        }
    }
}

// ---------------------------------------------------------------------------
// Steps 4(end)+5: node_energy = h2[n,:300] . W3 ; dg[batch[n]] += energy.
// One wave32 per node, shuffle reduction, one atomic per node.
// ---------------------------------------------------------------------------
__global__ void energy_pool(const float* __restrict__ h2,
                            const float* __restrict__ W3,
                            const int* __restrict__ batch,
                            float* __restrict__ dg, int N) {
    int lane = threadIdx.x & 31;
    int n = blockIdx.x * 8 + (threadIdx.x >> 5);
    if (n >= N) return;
    const float* row = h2 + (size_t)n * 300;
    float s = 0.0f;
    for (int f = lane; f < 300; f += 32) s += row[f] * W3[f];
    for (int off = 16; off > 0; off >>= 1) s += __shfl_down(s, off, 32);
    if (lane == 0) atomicAdd(&dg[batch[n]], s);
}

// ---------------------------------------------------------------------------
// launch
// ---------------------------------------------------------------------------
extern "C" void kernel_launch(void* const* d_in, const int* in_sizes, int n_in,
                              void* d_out, int out_size, void* d_ws, size_t ws_size,
                              hipStream_t stream) {
    (void)in_sizes; (void)n_in; (void)out_size; (void)ws_size;

    const int*   atom        = (const int*)d_in[0];
    const int*   bond_index  = (const int*)d_in[1];
    const int*   bond_attr   = (const int*)d_in[2];
    const int*   batch       = (const int*)d_in[3];
    const float* W_atom_num  = (const float*)d_in[4];
    const float* W_hyb       = (const float*)d_in[5];
    const float* W_arom      = (const float*)d_in[6];
    const float* W_chir      = (const float*)d_in[7];
    const float* W_charge    = (const float*)d_in[8];
    const float* W_bond_type = (const float*)d_in[9];
    const float* W_bond_at_i = (const float*)d_in[10];
    const float* W_sl_atom   = (const float*)d_in[11];
    const float* W_sl        = (const float*)d_in[12];
    const float* W1          = (const float*)d_in[13];
    const float* b1          = (const float*)d_in[14];
    const float* W2          = (const float*)d_in[15];
    const float* b2          = (const float*)d_in[16];
    const float* W3          = (const float*)d_in[17];
    float* dg = (float*)d_out;

    // workspace layout
    const size_t nodeBytes = (size_t)NND * EMB * sizeof(float);   // 120 MB
    float*          nodeA = (float*)d_ws;
    float*          nodeB = (float*)((char*)d_ws + nodeBytes);
    unsigned short* W1p   = (unsigned short*)((char*)d_ws + 2 * nodeBytes);
    unsigned short* W2p   = W1p + (size_t)KPAD * KPAD;            // 608 rows * 608 k

    const int NT1 = (EMB + 15) / 16;          // 38 N-tiles (padded rows = 608)
    const int NT2 = (EMB / 2 + 15) / 16;      // 19 N-tiles (padded rows = 304)
    const int MT  = NND / 16;                 // 3125 M-tiles (exact)

    // pack weights to padded bf16 (tiny; once per launch)
    pack_weights<<<(KPAD * KPAD + 255) / 256, 256, 0, stream>>>(W1, EMB, EMB, W1p, NT1 * 16, KPAD);
    pack_weights<<<(NT2 * 16 * KPAD + 255) / 256, 256, 0, stream>>>(W2, EMB / 2, EMB, W2p, NT2 * 16, KPAD);

    // Step 1: node embeddings -> nodeA
    node_embed<<<NND, 256, 0, stream>>>(atom, W_atom_num, W_hyb, W_arom, W_chir, W_charge, nodeA);

    // Steps 2+3: two MP layers (ping-pong), scatter-add targets zeroed first
    fill_f32<<<2048, 256, 0, stream>>>(nodeB, (long long)NND * EMB, 0.0f);
    mp_layer<<<NED + NND, 256, 0, stream>>>(nodeA, nodeB, bond_index, bond_attr, atom,
                                            W_bond_type, W_bond_at_i, W_sl_atom, W_sl);
    fill_f32<<<2048, 256, 0, stream>>>(nodeA, (long long)NND * EMB, 0.0f);
    mp_layer<<<NED + NND, 256, 0, stream>>>(nodeB, nodeA, bond_index, bond_attr, atom,
                                            W_bond_type, W_bond_at_i, W_sl_atom, W_sl);

    // Step 4: h1 = relu(relu(node) @ W1^T + b1)   [nodeA -> nodeB]
    gemm_bias_relu<5><<<MT, 256, 0, stream>>>(nodeA, NND, EMB, W1p, b1, nodeB, EMB, NT1, 1);
    //         h2 = relu(h1 @ W2^T + b2)           [nodeB -> nodeA as [N,300]]
    gemm_bias_relu<3><<<MT, 256, 0, stream>>>(nodeB, NND, EMB, W2p, b2, nodeA, EMB / 2, NT2, 0);

    // Step 5: energy + global add pool
    fill_f32<<<8, 128, 0, stream>>>(dg, NGR, 0.0f);
    energy_pool<<<(NND + 7) / 8, 256, 0, stream>>>(nodeA, W3, batch, dg, NND);
}
